// ASS_JRG_3186865733952
// MI455X (gfx1250) — compile-verified, matrix operand order checked by reading, not run on hardware
//
#include <hip/hip_runtime.h>
#include <hip/hip_bf16.h>

// ---------------------------------------------------------------------------
// ASS_JRG default config: B=128, T=10, J=8, M=5, D=128, WHOLE=1024, H=512
// rows = B*T*J*M = 51200, K = 1024, N = 128 per encoder.
// ---------------------------------------------------------------------------

typedef __attribute__((ext_vector_type(8)))  float  v8f;
typedef __attribute__((ext_vector_type(16))) __bf16 v16bf;

union FragAB {
    v16bf v;
    unsigned short s[16];
    unsigned u[8];
    uint4 q[2];
};

__device__ __forceinline__ unsigned short f2bf(float f) {
    unsigned int u = __float_as_uint(f);
    unsigned int lsb = (u >> 16) & 1u;
    u += 0x7fffu + lsb;                 // exact round-to-nearest-even
    return (unsigned short)(u >> 16);
}

// Packed f32x2 -> bf16x2 (nearest rounding via +0x8000, packed with V_PERM_B32)
__device__ __forceinline__ unsigned pk2bf(float lo, float hi) {
    unsigned ul = __float_as_uint(lo) + 0x8000u;
    unsigned uh = __float_as_uint(hi) + 0x8000u;
    return __builtin_amdgcn_perm(uh, ul, 0x07060302u);  // {uh[31:16], ul[31:16]}
}

__device__ __forceinline__ void wait_async0() {
#if __has_builtin(__builtin_amdgcn_s_wait_asynccnt)
    __builtin_amdgcn_s_wait_asynccnt(0);
#else
    asm volatile("s_wait_asynccnt 0x0" ::: "memory");
#endif
}

// Async DMA: 16 bytes/lane from global (saddr base + 32-bit voffset) into LDS.
__device__ __forceinline__ void async_b128_to_lds(unsigned lds_addr,
                                                  const void* sbase,
                                                  unsigned voff) {
    asm volatile("global_load_async_to_lds_b128 %0, %1, %2 offset:0"
                 :: "v"(lds_addr), "v"(voff), "s"(sbase)
                 : "memory");
}

// ---------------------------------------------------------------------------
// Kernel 0: pack W_whole / W_diff (1024x128 f32) into WMMA-B fragment order,
// bf16.  Layout: [ktile 0..31][ntile 0..7][lane 0..31][16 bf16 contig].
// lane element e -> B[k][n], k = kt*32 + (lane>=16?8:0) + (e>=8?16:0) + (e&7),
//                  n = nt*16 + (lane&15).   One K-step slab = 8 KB contiguous.
// ---------------------------------------------------------------------------
__global__ void pack_weights(const float* __restrict__ Ww,
                             const float* __restrict__ Wd,
                             unsigned short* __restrict__ dstW,
                             unsigned short* __restrict__ dstD) {
    int tid = blockIdx.x * blockDim.x + threadIdx.x;    // 16384 lane-tasks
    if (tid >= 2 * 32 * 8 * 32) return;
    int lane = tid & 31;
    int nt   = (tid >> 5) & 7;
    int kt   = (tid >> 8) & 31;
    int enc  = tid >> 13;

    const float* W = enc ? Wd : Ww;
    unsigned short* dst = enc ? dstD : dstW;

    int n     = nt * 16 + (lane & 15);
    int khalf = (lane >> 4) * 8;

    unsigned short tmp[16];
#pragma unroll
    for (int e = 0; e < 16; ++e) {
        int k = kt * 32 + khalf + ((e >= 8) ? 16 : 0) + (e & 7);
        tmp[e] = f2bf(W[(size_t)k * 128 + n]);
    }
    unsigned short* p = dst + (((size_t)(kt * 8 + nt)) * 32 + lane) * 16;
#pragma unroll
    for (int e = 0; e < 16; ++e) p[e] = tmp[e];
}

// ---------------------------------------------------------------------------
// Kernel 1: dual-encoder GEMM with fused temporal diff.
//   ew = relu(feat @ Ww + bw), ed = relu(|shift(feat)-feat| @ Wd + bd)
// 4 waves/block, one wave = 16 rows x 128 cols for BOTH encoders.
// B slabs (16 KB per K-step: [encW 8KB][encD 8KB]) are double-buffered in LDS
// via GLOBAL_LOAD_ASYNC_TO_LDS_B128, shared by all 4 waves.
// ---------------------------------------------------------------------------
__global__ __launch_bounds__(128) void enc_gemm(
        const float* __restrict__ feat,
        const unsigned short* __restrict__ wbW,
        const unsigned short* __restrict__ wbD,
        const float* __restrict__ biasW,
        const float* __restrict__ biasD,
        float* __restrict__ ew,
        float* __restrict__ ed) {
    __shared__ unsigned char bbuf[2 * 16384];

    const int tid     = threadIdx.x;
    const int lane    = tid & 31;
    const int wave    = tid >> 5;
    const int rowtile = blockIdx.x * 4 + wave;       // 0..3199
    const int mrow    = lane & 15;                   // row within 16-row tile
    const int hi      = lane >> 4;                   // K-half select (A layout)
    const int r       = rowtile * 16 + mrow;         // global row 0..51199
    const int sidx    = r % 400;                     // position within sequence
    const int rn      = (sidx == 399) ? r : r + 1;   // last-frame replication

    const float* rowc = feat + (size_t)r  * 1024;
    const float* rown = feat + (size_t)rn * 1024;

    const unsigned ldsbase = (unsigned)(size_t)(&bbuf[0]);

    // prologue: issue async batch for kt = 0 into buffer 0
    {
        const unsigned gofs = tid * 16;              // within 8 KB slab
#pragma unroll
        for (int i = 0; i < 4; ++i) {
            async_b128_to_lds(ldsbase +        i * 2048 + tid * 16, wbW, gofs + i * 2048);
            async_b128_to_lds(ldsbase + 8192 + i * 2048 + tid * 16, wbD, gofs + i * 2048);
        }
    }

    v8f accw[8] = {};
    v8f accd[8] = {};

    for (int kt = 0; kt < 32; ++kt) {
        const int cur = kt & 1;
        const int kbase = kt * 32 + hi * 8;

        // A-side global loads (independent of LDS traffic)
        float4 c0 = *(const float4*)(rowc + kbase);
        float4 c1 = *(const float4*)(rowc + kbase + 4);
        float4 c2 = *(const float4*)(rowc + kbase + 16);
        float4 c3 = *(const float4*)(rowc + kbase + 20);
        float4 n0 = *(const float4*)(rown + kbase);
        float4 n1 = *(const float4*)(rown + kbase + 4);
        float4 n2 = *(const float4*)(rown + kbase + 16);
        float4 n3 = *(const float4*)(rown + kbase + 20);

        // current B slab has landed; everyone past their reads of the other buf
        wait_async0();
        __syncthreads();

        // prefetch next K-step slab into the other buffer
        if (kt + 1 < 32) {
            const unsigned gofs = (unsigned)(kt + 1) * 8192 + tid * 16;
            const unsigned lofs = ldsbase + (cur ^ 1) * 16384 + tid * 16;
#pragma unroll
            for (int i = 0; i < 4; ++i) {
                async_b128_to_lds(lofs +        i * 2048, wbW, gofs + i * 2048);
                async_b128_to_lds(lofs + 8192 + i * 2048, wbD, gofs + i * 2048);
            }
        }

        // build bf16 A fragments: packed +0x8000 round + V_PERM pack
        FragAB aw, ad;
        const float cf[16] = {c0.x,c0.y,c0.z,c0.w, c1.x,c1.y,c1.z,c1.w,
                              c2.x,c2.y,c2.z,c2.w, c3.x,c3.y,c3.z,c3.w};
        const float nf[16] = {n0.x,n0.y,n0.z,n0.w, n1.x,n1.y,n1.z,n1.w,
                              n2.x,n2.y,n2.z,n2.w, n3.x,n3.y,n3.z,n3.w};
#pragma unroll
        for (int e = 0; e < 8; ++e) {
            aw.u[e] = pk2bf(cf[2 * e], cf[2 * e + 1]);
            ad.u[e] = pk2bf(fabsf(nf[2 * e]     - cf[2 * e]),
                            fabsf(nf[2 * e + 1] - cf[2 * e + 1]));
        }

        // B fragments from LDS, software-pipelined one nt ahead
        const unsigned char* bufp = bbuf + cur * 16384;
        FragAB bw0, bd0, bw1, bd1;
        {
            const uint4* pw = (const uint4*)(bufp + lane * 32);
            const uint4* pd = (const uint4*)(bufp + 8192 + lane * 32);
            bw0.q[0] = pw[0]; bw0.q[1] = pw[1];
            bd0.q[0] = pd[0]; bd0.q[1] = pd[1];
        }
#pragma unroll
        for (int nt = 0; nt < 8; ++nt) {
            if (nt + 1 < 8) {
                const uint4* pw = (const uint4*)(bufp + (nt + 1) * 1024 + lane * 32);
                const uint4* pd = (const uint4*)(bufp + 8192 + (nt + 1) * 1024 + lane * 32);
                bw1.q[0] = pw[0]; bw1.q[1] = pw[1];
                bd1.q[0] = pd[0]; bd1.q[1] = pd[1];
            }
            accw[nt] = __builtin_amdgcn_wmma_f32_16x16x32_bf16(
                false, aw.v, false, bw0.v, (short)0, accw[nt], false, false);
            accd[nt] = __builtin_amdgcn_wmma_f32_16x16x32_bf16(
                false, ad.v, false, bd0.v, (short)0, accd[nt], false, false);
            bw0 = bw1; bd0 = bd1;
        }
    }

    // C/D layout: lane<16 -> M = v, lane>=16 -> M = v+8 ; N = lane&15 + nt*16
#pragma unroll
    for (int nt = 0; nt < 8; ++nt) {
        const int col = nt * 16 + (lane & 15);
        const float bw = biasW[col];
        const float bd = biasD[col];
#pragma unroll
        for (int v = 0; v < 8; ++v) {
            const size_t row = (size_t)rowtile * 16 + hi * 8 + v;
            float vw = accw[nt][v] + bw; vw = vw > 0.f ? vw : 0.f;
            float vd = accd[nt][v] + bd; vd = vd > 0.f ? vd : 0.f;
            ew[row * 128 + col] = vw;
            ed[row * 128 + col] = vd;
        }
    }
}

// ---------------------------------------------------------------------------
// Kernel 2: graph message passing over joints + concat -> cell0 (B,T,J,M,512)
// ---------------------------------------------------------------------------
__global__ __launch_bounds__(256) void graph_concat(
        const float* __restrict__ ew,
        const float* __restrict__ ed,
        const float* __restrict__ gtm,
        float* __restrict__ cell0) {
    __shared__ float g[64];
    if (threadIdx.x < 64) {
        int j = threadIdx.x >> 3, k = threadIdx.x & 7;
        int dj = j - k;
        float v = gtm[threadIdx.x];
        g[threadIdx.x] = (dj <= 1 && dj >= -1) ? fabsf(v) : 0.f;
    }
    __syncthreads();

    const int idx = blockIdx.x * blockDim.x + threadIdx.x;   // 819200 exact
    const int d  = idx & 127;
    const int m  = (idx >> 7) % 5;
    const int bt = idx / (128 * 5);                          // b*10 + t

    const size_t rb = (size_t)bt * 40 + m;                   // row base (j=0)
    float ewj[8], edj[8];
#pragma unroll
    for (int j = 0; j < 8; ++j) {
        const size_t rr = rb + (size_t)j * 5;
        ewj[j] = ew[rr * 128 + d];
        edj[j] = ed[rr * 128 + d];
    }
#pragma unroll
    for (int k = 0; k < 8; ++k) {
        float pw = 0.f, pd = 0.f;
#pragma unroll
        for (int j = 0; j < 8; ++j) {
            const float gv = g[j * 8 + k];
            pw = fmaf(ewj[j], gv, pw);
            pd = fmaf(edj[j], gv, pd);
        }
        const size_t o = (((size_t)bt * 8 + k) * 5 + m) * 512;
        cell0[o +       d] = pw;       // ewp
        cell0[o + 128 + d] = pd;       // edp
        cell0[o + 256 + d] = ewj[k];   // ew
        cell0[o + 384 + d] = edj[k];   // ed
    }
}

// ---------------------------------------------------------------------------
// Kernel 3: fused = mean over (T,J,M)=400 of cell0 -> (B,512)
// ---------------------------------------------------------------------------
__global__ __launch_bounds__(256) void mean_tjm(
        const float* __restrict__ cell0, float* __restrict__ fused) {
    const int idx = blockIdx.x * blockDim.x + threadIdx.x;   // 65536 exact
    const int b = idx >> 9, c = idx & 511;
    const float* base = cell0 + (size_t)b * 400 * 512 + c;
    float s = 0.f;
    for (int i = 0; i < 400; ++i) s += base[(size_t)i * 512];
    fused[idx] = s * (1.f / 400.f);
}

// ---------------------------------------------------------------------------
// Kernel 4: fused2 = relu(fused) @ W_a1 + b_a1  -> (B,512)
// ---------------------------------------------------------------------------
__global__ __launch_bounds__(256) void head(
        const float* __restrict__ fused,
        const float* __restrict__ Wa1,
        const float* __restrict__ ba1,
        float* __restrict__ out) {
    const int idx = blockIdx.x * blockDim.x + threadIdx.x;   // 65536 exact
    const int b = idx >> 9, h = idx & 511;
    const float* f = fused + (size_t)b * 512;
    float acc = ba1[h];
    for (int c = 0; c < 512; ++c) {
        float x = f[c]; x = x > 0.f ? x : 0.f;
        acc = fmaf(x, Wa1[(size_t)c * 512 + h], acc);
    }
    out[idx] = acc;
}

// ---------------------------------------------------------------------------
extern "C" void kernel_launch(void* const* d_in, const int* in_sizes, int n_in,
                              void* d_out, int out_size, void* d_ws, size_t ws_size,
                              hipStream_t stream) {
    const float* feat  = (const float*)d_in[0];   // (128, 400, 1024)
    const float* Ww    = (const float*)d_in[1];   // (1024, 128)
    const float* bw    = (const float*)d_in[2];   // (128,)
    const float* Wd    = (const float*)d_in[3];   // (1024, 128)
    const float* bd    = (const float*)d_in[4];   // (128,)
    const float* gtm   = (const float*)d_in[5];   // (1, 8, 8)
    const float* Wa1   = (const float*)d_in[6];   // (512, 512)
    const float* ba1   = (const float*)d_in[7];   // (512,)

    float* out   = (float*)d_out;                 // fused2 (65536) ++ cell0
    float* cell0 = out + 65536;

    // workspace layout (bytes)
    char* ws = (char*)d_ws;
    unsigned short* wbW = (unsigned short*)(ws);                 // 256 KB
    unsigned short* wbD = (unsigned short*)(ws + 262144);        // 256 KB
    float* ew    = (float*)(ws + 524288);                        // 25 MB
    float* ed    = (float*)(ws + 524288 + 26214400);             // 25 MB
    float* fused = (float*)(ws + 524288 + 2 * 26214400);         // 256 KB

    // 0) weight pack fp32 -> bf16 fragment order
    pack_weights<<<64, 256, 0, stream>>>(Ww, Wd, wbW, wbD);
    // 1) dual encoder GEMM (3200 wave tiles, 4 waves/block)
    enc_gemm<<<800, 128, 0, stream>>>(feat, wbW, wbD, bw, bd, ew, ed);
    // 2) graph message passing + concat -> cell0
    graph_concat<<<3200, 256, 0, stream>>>(ew, ed, gtm, cell0);
    // 3) mean over (T,J,M)
    mean_tjm<<<256, 256, 0, stream>>>(cell0, fused);
    // 4) head
    head<<<256, 256, 0, stream>>>(fused, Wa1, ba1, out);
}
